// CustomRNNCell_37048387895679
// MI455X (gfx1250) — matrix operand order repaired and checked by reading, not compiled, not run on hardware
//
#include <hip/hip_runtime.h>
#include <math.h>

typedef __bf16 bf16_t;
typedef __attribute__((ext_vector_type(16))) __bf16 v16bf;
typedef __attribute__((ext_vector_type(8)))  __bf16 v8bf;
typedef __attribute__((ext_vector_type(8)))  float  v8f;

constexpr int HID   = 1024;   // hidden dim == input dim
constexpr int BATCH = 64;
constexpr int SEQ   = 512;
constexpr int WSTR  = 1032;   // LDS row stride, resident weights (1024+8 pad, 16B-multiple)
constexpr int TSTR  = 80;     // LDS row stride, staged 64-wide tiles (64+16 pad)

// ---------------- fragment builders (ISA §7.12.2 layouts) ----------------

// A: 16x32 bf16 fragment from row-major bf16 storage (global or LDS).
// lanes 0-15 = rows; K halves {kb..kb+7, kb+16..kb+23}, kb = 0 / 8 by lane half.
__device__ __forceinline__ v16bf load_frag_a_bf16(const bf16_t* g, long row_stride,
                                                  int mrow, int k, int lane) {
    const int row   = mrow + (lane & 15);
    const int kbase = k + ((lane & 16) ? 8 : 0);
    const bf16_t* p = g + (long)row * row_stride + kbase;
    v8bf lo = *(const v8bf*)(p);
    v8bf hi = *(const v8bf*)(p + 16);
    return __builtin_shufflevector(lo, hi, 0,1,2,3,4,5,6,7,8,9,10,11,12,13,14,15);
}

// A fragment loaded from row-major f32 global memory, converted to bf16 in regs.
__device__ __forceinline__ v16bf load_frag_a_f32(const float* __restrict__ g,
                                                 long row_stride, int mrow,
                                                 int k, int lane) {
    const int row   = mrow + (lane & 15);
    const int kbase = k + ((lane & 16) ? 8 : 0);
    const float* p  = g + (long)row * row_stride + kbase;
    float4 f0 = *(const float4*)(p);
    float4 f1 = *(const float4*)(p + 4);
    float4 f2 = *(const float4*)(p + 16);
    float4 f3 = *(const float4*)(p + 20);
    v16bf a;
    a[0] =(__bf16)f0.x; a[1] =(__bf16)f0.y; a[2] =(__bf16)f0.z; a[3] =(__bf16)f0.w;
    a[4] =(__bf16)f1.x; a[5] =(__bf16)f1.y; a[6] =(__bf16)f1.z; a[7] =(__bf16)f1.w;
    a[8] =(__bf16)f2.x; a[9] =(__bf16)f2.y; a[10]=(__bf16)f2.z; a[11]=(__bf16)f2.w;
    a[12]=(__bf16)f3.x; a[13]=(__bf16)f3.y; a[14]=(__bf16)f3.z; a[15]=(__bf16)f3.w;
    return a;
}

// B: 32x16 bf16 fragment from row-major [n][k] bf16 LDS (computing X @ W^T).
// lane&15 = N column; K halves {kb..kb+7, kb+16..kb+23}, kb = 0 / 8 by lane half.
__device__ __forceinline__ v16bf load_frag_b_lds(const bf16_t* s, int row_stride,
                                                 int nrow, int k, int lane) {
    const int row   = nrow + (lane & 15);
    const int kbase = k + ((lane & 16) ? 8 : 0);
    const bf16_t* p = s + row * row_stride + kbase;
    v8bf lo = *(const v8bf*)(p);
    v8bf hi = *(const v8bf*)(p + 16);
    return __builtin_shufflevector(lo, hi, 0,1,2,3,4,5,6,7,8,9,10,11,12,13,14,15);
}

// ---------------- LDS staging for the xp GEMM ----------------

__device__ __forceinline__ void stage_f32_tile(bf16_t* s, const float* __restrict__ g,
                                               long row_stride_g, int k0, int tid) {
    int r  = tid >> 2;
    int cb = (tid & 3) << 4;
    const float4* p = (const float4*)(g + (long)r * row_stride_g + k0 + cb);
    float4 f0 = p[0], f1 = p[1], f2 = p[2], f3 = p[3];
    v8bf lo, hi;
    lo[0]=(__bf16)f0.x; lo[1]=(__bf16)f0.y; lo[2]=(__bf16)f0.z; lo[3]=(__bf16)f0.w;
    lo[4]=(__bf16)f1.x; lo[5]=(__bf16)f1.y; lo[6]=(__bf16)f1.z; lo[7]=(__bf16)f1.w;
    hi[0]=(__bf16)f2.x; hi[1]=(__bf16)f2.y; hi[2]=(__bf16)f2.z; hi[3]=(__bf16)f2.w;
    hi[4]=(__bf16)f3.x; hi[5]=(__bf16)f3.y; hi[6]=(__bf16)f3.z; hi[7]=(__bf16)f3.w;
    *(v8bf*)(s + r * TSTR + cb)     = lo;
    *(v8bf*)(s + r * TSTR + cb + 8) = hi;
}

__device__ __forceinline__ void stage_bf16_tile(bf16_t* s, const bf16_t* __restrict__ g,
                                                int n0, int k0, int tid) {
    int r  = tid >> 2;
    int cb = (tid & 3) << 4;
    const uint4* p = (const uint4*)(g + (long)(n0 + r) * HID + k0 + cb);
    uint4 w0 = p[0], w1 = p[1];
    *(uint4*)(s + r * TSTR + cb)     = w0;
    *(uint4*)(s + r * TSTR + cb + 8) = w1;
}

// ---------------- grid barrier helper (device-scope monotonic counter) ----------------

__device__ __forceinline__ void grid_barrier(unsigned int* ctr, int tid,
                                             unsigned int target) {
    __syncthreads();
    if (tid == 0) {
        __threadfence();   // publish this WG's stores
        __hip_atomic_fetch_add(ctr, 1u, __ATOMIC_RELEASE, __HIP_MEMORY_SCOPE_AGENT);
        while (__hip_atomic_load(ctr, __ATOMIC_ACQUIRE, __HIP_MEMORY_SCOPE_AGENT) < target) {
            __builtin_amdgcn_s_sleep(1);
        }
    }
    __syncthreads();
    __threadfence();       // acquire other WGs' stores
}

// ================= PATH A: one-shot xp GEMM + light scan =================

// xp[t][b][:] = x[b][t][:] @ Wx^T + (b_h + b_x).  M = B*SEQ, fully parallel.
__global__ __launch_bounds__(256)
void xp_gemm_kernel(const float* __restrict__ x_seq,   // [B*SEQ][HID] (flat row = b*SEQ+t)
                    const bf16_t* __restrict__ Wx,     // [HID][HID] bf16
                    const float* __restrict__ bias,    // b_h + b_x
                    float* __restrict__ xp) {          // [SEQ][B][HID] f32
    __shared__ __attribute__((aligned(16))) bf16_t sA[64 * TSTR];
    __shared__ __attribute__((aligned(16))) bf16_t sB[64 * TSTR];

    const int tid   = threadIdx.x;
    const int lane  = tid & 31;
    const int wave  = tid >> 5;
    const int mIdx  = wave & 3;
    const int nPair = wave >> 2;
    const int m0    = blockIdx.x * 64;   // flat x row base
    const int n0    = blockIdx.y * 64;

    v8f acc0 = {}, acc1 = {};
    const int nr0 = (2 * nPair) * 16;
    const int nr1 = nr0 + 16;

    for (int k0 = 0; k0 < HID; k0 += 64) {
        __syncthreads();
        stage_f32_tile(sA, x_seq + (long)m0 * HID, HID, k0, tid);
        stage_bf16_tile(sB, Wx, n0, k0, tid);
        __syncthreads();
#pragma unroll
        for (int kk = 0; kk < 64; kk += 32) {
            v16bf a  = load_frag_a_bf16(sA, TSTR, mIdx * 16, kk, lane);
            v16bf b0 = load_frag_b_lds(sB, TSTR, nr0, kk, lane);
            v16bf b1 = load_frag_b_lds(sB, TSTR, nr1, kk, lane);
            acc0 = __builtin_amdgcn_wmma_f32_16x16x32_bf16(false, a, false, b0, (short)0, acc0, false, false);
            acc1 = __builtin_amdgcn_wmma_f32_16x16x32_bf16(false, a, false, b1, (short)0, acc1, false, false);
        }
    }

    const int colL0 = n0 + nr0 + (lane & 15);
    const int colL1 = colL0 + 16;
    const float bv0 = bias[colL0];
    const float bv1 = bias[colL1];
    const int rowbase = mIdx * 16 + ((lane & 16) ? 8 : 0);
#pragma unroll
    for (int r = 0; r < 8; ++r) {
        const int frow = m0 + rowbase + r;            // flat = b*SEQ + t
        const int b = frow >> 9;                      // / SEQ
        const int t = frow & (SEQ - 1);
        float* dst = xp + ((long)t * BATCH + b) * HID;
        dst[colL0] = acc0[r] + bv0;
        dst[colL1] = acc1[r] + bv1;
    }
}

// Scan with precomputed xp: h = tanh(h @ Wh^T + xp_t).
// 32 WGs x 32-col slices; only a 64.5 KB Wh slice resident per WG.
// 8 waves x 1 tile each -> 32-deep WMMA chain per step (halved vs 16-WG config).
__global__ __launch_bounds__(256)
void rnn_scan_xp_kernel(const float*  __restrict__ xp,   // [SEQ][B][HID]
                        const bf16_t* __restrict__ Wh,
                        bf16_t* __restrict__ hbuf0,      // [B][HID] bf16 (zeroed)
                        bf16_t* __restrict__ hbuf1,
                        float* __restrict__ out,
                        unsigned int* __restrict__ ctr) {
    __shared__ __attribute__((aligned(16))) bf16_t sWh[32 * WSTR];

    const int tid  = threadIdx.x;
    const int lane = tid & 31;
    const int wave = tid >> 5;
    const int mIdx = wave & 3;            // 16-row block of the 64 batch rows
    const int nIdx = wave >> 2;           // 16-col block within the 32-col slice
    const int n0   = blockIdx.x * 32;

    {   // one-time Wh slice preload: 8 threads per row, 256B chunks
        const int row = tid >> 3;                    // 0..31
        const int cb  = (tid & 7) * 128;             // bf16 column base
        const uint4* gh = (const uint4*)(Wh + (long)(n0 + row) * HID + cb);
        uint4* lh = (uint4*)(sWh + row * WSTR + cb);
#pragma unroll
        for (int i = 0; i < 16; ++i) lh[i] = gh[i];  // 16 x 16B = 256B
    }
    __syncthreads();

    const int nr   = nIdx * 16;
    const int colG = n0 + nr + (lane & 15);
    const int rowbase = mIdx * 16 + ((lane & 16) ? 8 : 0);
    const unsigned int nwg = gridDim.x;

    for (int t = 0; t < SEQ; ++t) {
        const bf16_t* hin  = (t & 1) ? hbuf1 : hbuf0;
        bf16_t*       hout = (t & 1) ? hbuf0 : hbuf1;
        const float*  xpt  = xp + (long)t * BATCH * HID;

        v8f acc = {};
#pragma unroll 4
        for (int k = 0; k < HID; k += 32) {
            v16bf aH = load_frag_a_bf16(hin, HID, mIdx * 16, k, lane);
            v16bf b  = load_frag_b_lds(sWh, WSTR, nr, k, lane);
            acc = __builtin_amdgcn_wmma_f32_16x16x32_bf16(false, aH, false, b, (short)0, acc, false, false);
        }

#pragma unroll
        for (int r = 0; r < 8; ++r) {
            const int row = rowbase + r;
            const float v = tanhf(acc[r] + xpt[(long)row * HID + colG]);
            if (t == SEQ - 1) out[(long)row * HID + colG] = v;
            else              hout[(long)row * HID + colG] = (__bf16)v;
        }
        grid_barrier(ctr, tid, nwg * (unsigned int)(t + 1));
    }
}

// ================= PATH B: fused scan (small workspace) =================
// 32 WGs x 32-col slices; Wh+Wx slices resident (129 KB LDS per WG).

__global__ __launch_bounds__(256)
void rnn_scan_fused_kernel(const float*  __restrict__ x_seq,
                           const bf16_t* __restrict__ Wh,
                           const bf16_t* __restrict__ Wx,
                           const float*  __restrict__ bias,
                           bf16_t* __restrict__ hbuf0,
                           bf16_t* __restrict__ hbuf1,
                           float* __restrict__ out,
                           unsigned int* __restrict__ ctr) {
    __shared__ __attribute__((aligned(16))) bf16_t sWh[32 * WSTR];
    __shared__ __attribute__((aligned(16))) bf16_t sWx[32 * WSTR];

    const int tid  = threadIdx.x;
    const int lane = tid & 31;
    const int wave = tid >> 5;
    const int mIdx = wave & 3;
    const int nIdx = wave >> 2;
    const int n0   = blockIdx.x * 32;

    {
        const int row = tid >> 3;
        const int cb  = (tid & 7) * 128;
        const uint4* gh = (const uint4*)(Wh + (long)(n0 + row) * HID + cb);
        const uint4* gx = (const uint4*)(Wx + (long)(n0 + row) * HID + cb);
        uint4* lh = (uint4*)(sWh + row * WSTR + cb);
        uint4* lx = (uint4*)(sWx + row * WSTR + cb);
#pragma unroll
        for (int i = 0; i < 16; ++i) { lh[i] = gh[i]; lx[i] = gx[i]; }
    }
    __syncthreads();

    const int nr   = nIdx * 16;
    const int colG = n0 + nr + (lane & 15);
    const float bv = bias[colG];
    const int rowbase = mIdx * 16 + ((lane & 16) ? 8 : 0);
    const unsigned int nwg = gridDim.x;

    for (int t = 0; t < SEQ; ++t) {
        const bf16_t* hin  = (t & 1) ? hbuf1 : hbuf0;
        bf16_t*       hout = (t & 1) ? hbuf0 : hbuf1;
        const float*  xt   = x_seq + (long)t * HID;

        v8f acc = {};
#pragma unroll 2
        for (int k = 0; k < HID; k += 32) {
            v16bf aH = load_frag_a_bf16(hin, HID, mIdx * 16, k, lane);
            v16bf aX = load_frag_a_f32(xt, (long)SEQ * HID, mIdx * 16, k, lane);
            v16bf bH = load_frag_b_lds(sWh, WSTR, nr, k, lane);
            v16bf bX = load_frag_b_lds(sWx, WSTR, nr, k, lane);
            acc = __builtin_amdgcn_wmma_f32_16x16x32_bf16(false, aH, false, bH, (short)0, acc, false, false);
            acc = __builtin_amdgcn_wmma_f32_16x16x32_bf16(false, aX, false, bX, (short)0, acc, false, false);
        }

#pragma unroll
        for (int r = 0; r < 8; ++r) {
            const int row = rowbase + r;
            const float v = tanhf(acc[r] + bv);
            if (t == SEQ - 1) out[(long)row * HID + colG] = v;
            else              hout[(long)row * HID + colG] = (__bf16)v;
        }
        grid_barrier(ctr, tid, nwg * (unsigned int)(t + 1));
    }
}

// ---------------- setup kernels ----------------

__global__ __launch_bounds__(256)
void convert_weights_kernel(const float* __restrict__ Wh_f, const float* __restrict__ Wx_f,
                            bf16_t* __restrict__ Wh_b, bf16_t* __restrict__ Wx_b) {
    int i = blockIdx.x * blockDim.x + threadIdx.x;
    Wh_b[i] = (__bf16)Wh_f[i];
    Wx_b[i] = (__bf16)Wx_f[i];
}

__global__ __launch_bounds__(256)
void init_kernel(const float* __restrict__ b_h, const float* __restrict__ b_x,
                 float* __restrict__ bias, bf16_t* __restrict__ h0,
                 unsigned int* __restrict__ ctr) {
    int i = blockIdx.x * blockDim.x + threadIdx.x;   // BATCH*HID threads
    h0[i] = (__bf16)0.0f;                            // h_{-1} = 0
    if (i < HID) bias[i] = b_h[i] + b_x[i];
    if (i == 0) *ctr = 0u;
}

// ---------------- host launch ----------------

extern "C" void kernel_launch(void* const* d_in, const int* in_sizes, int n_in,
                              void* d_out, int out_size, void* d_ws, size_t ws_size,
                              hipStream_t stream) {
    (void)in_sizes; (void)n_in; (void)out_size;

    const float* x_seq = (const float*)d_in[0];
    const float* W_h   = (const float*)d_in[1];
    const float* b_h   = (const float*)d_in[2];
    const float* W_x   = (const float*)d_in[3];
    const float* b_x   = (const float*)d_in[4];
    float* out = (float*)d_out;

    // common workspace layout
    char* ws = (char*)d_ws;
    bf16_t* Wh_b  = (bf16_t*)ws;                                        // 2 MB
    bf16_t* Wx_b  = (bf16_t*)(ws + (size_t)HID * HID * 2);              // 2 MB
    float*  bias  = (float*)(ws + (size_t)HID * HID * 4);               // 4 KB
    bf16_t* hbuf0 = (bf16_t*)(ws + (size_t)HID * HID * 4 + 4096);       // 128 KB
    bf16_t* hbuf1 = hbuf0 + (size_t)BATCH * HID;                        // 128 KB
    unsigned int* ctr = (unsigned int*)(hbuf1 + (size_t)BATCH * HID);   // 4 B (+pad)
    float* xp = (float*)((char*)ctr + 256);                             // 134 MB (path A)

    const size_t base_need = (size_t)HID * HID * 4 + 4096
                           + (size_t)2 * BATCH * HID * 2 + 256;
    const size_t xp_bytes  = (size_t)SEQ * BATCH * HID * sizeof(float);
    const bool have_xp_ws  = ws_size >= base_need + xp_bytes;

    convert_weights_kernel<<<(HID * HID) / 256, 256, 0, stream>>>(W_h, W_x, Wh_b, Wx_b);
    init_kernel<<<(BATCH * HID) / 256, 256, 0, stream>>>(b_h, b_x, bias, hbuf0, ctr);

    if (have_xp_ws) {
        // Path A: hoist input projection off the critical path (fully parallel GEMM),
        // then a light persistent scan with only Wh resident.
        dim3 grid((BATCH * SEQ) / 64, HID / 64);
        xp_gemm_kernel<<<grid, 256, 0, stream>>>(x_seq, Wx_b, bias, xp);
        rnn_scan_xp_kernel<<<HID / 32, 256, 0, stream>>>(xp, Wh_b, hbuf0, hbuf1, out, ctr);
    } else {
        // Path B: fused scan (x projection computed inside each step).
        rnn_scan_fused_kernel<<<HID / 32, 256, 0, stream>>>(x_seq, Wh_b, Wx_b, bias,
                                                            hbuf0, hbuf1, out, ctr);
    }
}
